// ColorHistograms_37400575214078
// MI455X (gfx1250) — matrix operand order repaired and checked by reading, not compiled, not run on hardware
//
#include <hip/hip_runtime.h>
#include <hip/hip_bf16.h>

// ---------------- problem constants ----------------
#define BATCH   4
#define TFR     1024          // frames per batch
#define NPIX    1296          // 27*48 pixels per frame
#define NB      512           // histogram bins (= GEMM K)
#define LOOKUP  101
#define ODIM    128
#define PADROWS 64            // zero rows padded on each side of x
#define TPAD    (TFR + 2*PADROWS)   // 1152 rows per batch in padded x
#define NTILES  9             // 9 x 16 = 144 cols cover the 116-wide band

typedef __attribute__((ext_vector_type(16))) _Float16 v16h;
typedef __attribute__((ext_vector_type(8)))  float    v8f;
typedef __attribute__((ext_vector_type(4)))  float    v4f;

union V16 { v16h h; v4f f[2]; };

// ws layout (in _Float16 elements)
#define XPAD_HALFS ((size_t)BATCH * TPAD * NB)       // 2,359,296
#define WIN_HALFS  ((size_t)BATCH * TFR * ODIM)      //   524,288
#define WT_HALFS   ((size_t)ODIM * ODIM)             //    16,384

// ---------------------------------------------------------------------------
// Kernel 0: fc_w (101x128 f32, row-major [l][o]) -> transposed, K-padded f16
//           wT[o][k] = fc_w[k][o] for k<101, else 0   (128x128)
// ---------------------------------------------------------------------------
__global__ __launch_bounds__(256) void ch_wt_kernel(const float* __restrict__ fc_w,
                                                    _Float16* __restrict__ wT) {
    int idx = blockIdx.x * 256 + threadIdx.x;      // 0..16383
    int o = idx >> 7;
    int k = idx & 127;
    float v = (k < LOOKUP) ? fc_w[k * ODIM + o] : 0.0f;
    wT[o * ODIM + k] = (_Float16)v;
}

// ---------------------------------------------------------------------------
// Kernel 1: per-frame 512-bin histogram + L2 normalize -> f16 x_pad
// One block (256 threads) per frame; LDS float histogram + tree reduction.
// Bandwidth-dominant kernel (reads 64 MB of frames): 4 pixels = 12 dwords =
// three 16B-aligned int4 loads -> global_load_b128, then 4 LDS float atomics.
// ---------------------------------------------------------------------------
__global__ __launch_bounds__(256) void ch_hist_kernel(const int* __restrict__ frames,
                                                      _Float16* __restrict__ xpad) {
    __shared__ float hist[NB];
    __shared__ float red[256];
    const int tid = threadIdx.x;
    const int fi  = blockIdx.x;                    // 0 .. BATCH*TFR-1

    hist[tid]       = 0.0f;
    hist[tid + 256] = 0.0f;
    __syncthreads();

    // frame base: fi*NPIX*3 dwords = fi*15552 bytes (16B aligned);
    // chunk stride: 12 dwords = 48 bytes (16B aligned).
    const int4* fp = reinterpret_cast<const int4*>(frames + (size_t)fi * NPIX * 3);
    for (int c = tid; c < NPIX / 4; c += 256) {    // 324 chunks of 4 pixels
        int4 p0 = fp[c * 3 + 0];
        int4 p1 = fp[c * 3 + 1];
        int4 p2 = fp[c * 3 + 2];
        atomicAdd(&hist[((p0.x >> 5) << 6) + ((p0.y >> 5) << 3) + (p0.z >> 5)], 1.0f);
        atomicAdd(&hist[((p0.w >> 5) << 6) + ((p1.x >> 5) << 3) + (p1.y >> 5)], 1.0f);
        atomicAdd(&hist[((p1.z >> 5) << 6) + ((p1.w >> 5) << 3) + (p2.x >> 5)], 1.0f);
        atomicAdd(&hist[((p2.y >> 5) << 6) + ((p2.z >> 5) << 3) + (p2.w >> 5)], 1.0f);
    }
    __syncthreads();

    float h0 = hist[tid], h1 = hist[tid + 256];
    red[tid] = h0 * h0 + h1 * h1;
    __syncthreads();
    #pragma unroll
    for (int s = 128; s > 0; s >>= 1) {
        if (tid < s) red[tid] += red[tid + s];
        __syncthreads();
    }
    float inv = 1.0f / fmaxf(sqrtf(red[0]), 1e-12f);

    const int bb = fi >> 10;                       // fi / TFR
    const int t  = fi & (TFR - 1);
    _Float16* dst = xpad + ((size_t)bb * TPAD + PADROWS + t) * NB;
    dst[tid]       = (_Float16)(h0 * inv);
    dst[tid + 256] = (_Float16)(h1 * inv);
}

// ---------------------------------------------------------------------------
// Kernel 2: banded sims via WMMA.  One wave = one 16-row t-tile; 9 n-tiles of
// 16 cols cover j in [t0-64, t0+80) (needed band: [t0-50, t0+65]).
// A = x rows (16x512), B = x^T -> lane n reads ROW j0+n of x (no transpose!).
// Scatter D into win[b,t,l] with l = 16*nt + N - M - 14.
// ---------------------------------------------------------------------------
__global__ __launch_bounds__(256) void ch_sims_kernel(const _Float16* __restrict__ xpad,
                                                      _Float16* __restrict__ win) {
    const int lane = threadIdx.x & 31;
    const int wave = threadIdx.x >> 5;
    const int bb   = blockIdx.y;
    const int t0   = (blockIdx.x * 8 + wave) * 16;
    const int m    = lane & 15;                    // M-row (A) / N-col (B) index
    const int g    = lane >> 4;                    // lane half-group

    const _Float16* xb   = xpad + (size_t)bb * TPAD * NB;
    const _Float16* arow = xb + (size_t)(PADROWS + t0 + m) * NB;

    for (int nt = 0; nt < NTILES; ++nt) {
        // padded B row for column n=m of this n-tile: j0p = t0 + nt*16
        const _Float16* brow = xb + (size_t)(t0 + nt * 16 + m) * NB;
        v8f acc = {};
        #pragma unroll
        for (int ks = 0; ks < NB / 32; ++ks) {
            const int kb = ks * 32;
            V16 a, b;
            // A 16x32 f16 layout: lane g=0 -> K {kb..kb+7, kb+16..kb+23}
            //                     lane g=1 -> K {kb+8..kb+15, kb+24..kb+31}
            a.f[0] = *reinterpret_cast<const v4f*>(arow + kb + g * 8);
            a.f[1] = *reinterpret_cast<const v4f*>(arow + kb + 16 + g * 8);
            // B 32x16 f16 layout: lane (N=m,g) holds K {kb+g*8..+7, kb+16+g*8..+7} (same as A)
            b.f[0] = *reinterpret_cast<const v4f*>(brow + kb + g * 8);
            b.f[1] = *reinterpret_cast<const v4f*>(brow + kb + 16 + g * 8);
            acc = __builtin_amdgcn_wmma_f32_16x16x32_f16(
                      false, a.h, false, b.h, (short)0, acc, false, false);
        }
        // D layout: VGPR r -> M = r + 8*g ; N = m.  l = j - t + 50.
        #pragma unroll
        for (int r = 0; r < 8; ++r) {
            int M = r + 8 * g;
            int l = nt * 16 + m - M - 14;
            if (l >= 0 && l < LOOKUP) {
                win[((size_t)(bb * TFR + t0 + M)) * ODIM + l] = (_Float16)acc[r];
            }
        }
    }
}

// ---------------------------------------------------------------------------
// Kernel 3: out = relu(win @ fc_w + fc_b) via WMMA (K=128 with exact zero pad),
// bias folded into the accumulator init (bias depends only on N).
// ---------------------------------------------------------------------------
__global__ __launch_bounds__(256) void ch_fc_kernel(const _Float16* __restrict__ win,
                                                    const _Float16* __restrict__ wT,
                                                    const float* __restrict__ fc_b,
                                                    float* __restrict__ out) {
    const int lane = threadIdx.x & 31;
    const int wave = threadIdx.x >> 5;
    const int bb   = blockIdx.y;
    const int t0   = (blockIdx.x * 8 + wave) * 16;
    const int m    = lane & 15;
    const int g    = lane >> 4;

    const _Float16* arow = win + (size_t)(bb * TFR + t0 + m) * ODIM;

    for (int nt = 0; nt < ODIM / 16; ++nt) {
        const int o0 = nt * 16;
        const float bias = fc_b[o0 + m];           // N = m for all 8 D rows
        v8f acc = {bias, bias, bias, bias, bias, bias, bias, bias};
        const _Float16* brow = wT + (size_t)(o0 + m) * ODIM;
        #pragma unroll
        for (int ks = 0; ks < ODIM / 32; ++ks) {
            const int kb = ks * 32;
            V16 a, b;
            a.f[0] = *reinterpret_cast<const v4f*>(arow + kb + g * 8);
            a.f[1] = *reinterpret_cast<const v4f*>(arow + kb + 16 + g * 8);
            b.f[0] = *reinterpret_cast<const v4f*>(brow + kb + g * 8);
            b.f[1] = *reinterpret_cast<const v4f*>(brow + kb + 16 + g * 8);
            acc = __builtin_amdgcn_wmma_f32_16x16x32_f16(
                      false, a.h, false, b.h, (short)0, acc, false, false);
        }
        #pragma unroll
        for (int r = 0; r < 8; ++r) {
            int M = r + 8 * g;
            out[((size_t)(bb * TFR + t0 + M)) * ODIM + o0 + m] = fmaxf(acc[r], 0.0f);
        }
    }
}

// ---------------------------------------------------------------------------
extern "C" void kernel_launch(void* const* d_in, const int* in_sizes, int n_in,
                              void* d_out, int out_size, void* d_ws, size_t ws_size,
                              hipStream_t stream) {
    const int*   frames = (const int*)d_in[0];     // (4,1024,27,48,3) int32
    const float* fc_w   = (const float*)d_in[1];   // (101,128) f32
    const float* fc_b   = (const float*)d_in[2];   // (128,)    f32
    float*       out    = (float*)d_out;           // (4,1024,128) f32

    _Float16* xpad = (_Float16*)d_ws;
    _Float16* win  = xpad + XPAD_HALFS;
    _Float16* wT   = win + WIN_HALFS;

    // Zero x_pad (for the 64-row halo) and win (for l=101..127 K-padding).
    hipMemsetAsync(d_ws, 0, (XPAD_HALFS + WIN_HALFS) * sizeof(_Float16), stream);

    ch_wt_kernel  <<<dim3(WT_HALFS / 256), dim3(256), 0, stream>>>(fc_w, wT);
    ch_hist_kernel<<<dim3(BATCH * TFR),    dim3(256), 0, stream>>>(frames, xpad);
    ch_sims_kernel<<<dim3(TFR / 128, BATCH), dim3(256), 0, stream>>>(xpad, win);
    ch_fc_kernel  <<<dim3(TFR / 128, BATCH), dim3(256), 0, stream>>>(win, wT, fc_b, out);
}
